// PhysicsInformedEvolution_67903432949967
// MI455X (gfx1250) — matrix operand order repaired and checked by reading, not compiled, not run on hardware
//
#include <hip/hip_runtime.h>
#include <hip/hip_bf16.h>

typedef __bf16 bf16_t;
typedef __attribute__((ext_vector_type(16))) __bf16 v16bf;
typedef __attribute__((ext_vector_type(8)))  float  v8f;

#define BATCH 1024
#define HID   512
#define LAT   256
#define GDIM  2048   // 4*HID
#define TLEN  32
#define FFH   256    // HID/2

// ---------------------------------------------------------------- helpers
__device__ __forceinline__ float sigmoidf_(float x) {
    return 1.0f / (1.0f + __expf(-x));
}
__device__ __forceinline__ float gelu_tanh_(float x) {
    float x3 = x * x * x;
    return 0.5f * x * (1.0f + tanhf(0.7978845608f * (x + 0.044715f * x3)));
}

// Packed A-fragment index for element (m, k) of an M x K matrix (K = NKT*32).
// A-frag layout (16x32 bf16): lane&15 = m, kh=(lane>>4)*8;
//   e<8  -> k = kh + e ; e>=8 -> k = kh + 16 + (e-8)
__device__ __forceinline__ size_t packa_idx(int m, int k, int NKT) {
    int mt  = m >> 4;
    int kt  = k >> 5;
    int kin = k & 31;
    int hi  = (kin >> 3) & 1;
    int e   = (kin & 7) + ((kin >> 4) << 3);
    int lane = (m & 15) | (hi << 4);
    return (((size_t)mt * NKT + kt) * 32 + lane) * 16 + e;
}

// ---------------------------------------------------------------- setup kernels
__global__ void zero_f32_kernel(float* p, int n) {
    int i = blockIdx.x * blockDim.x + threadIdx.x;
    if (i < n) p[i] = 0.0f;
}

__global__ void init_kernel(const float* __restrict__ init_states,
                            const float* __restrict__ volt,
                            const float* __restrict__ thick,
                            const float* __restrict__ trap,
                            float* __restrict__ E_ws, float* __restrict__ E_out,
                            float* __restrict__ rate, float* __restrict__ s) {
    int b = blockIdx.x * blockDim.x + threadIdx.x;
    if (b >= BATCH) return;
    float E = volt[b] / thick[b];
    E_ws[b] = E;
    E_out[b] = E;
    float Ea = trap[b * 8 + 0], ga = trap[b * 8 + 1];
    float a = Ea - ga * E;
    a = a > 0.0f ? a : 0.0f;
    rate[b] = __expf(-a * (1.0f / 0.025851f));   // KB_T = 8.617e-5*300
    s[b] = init_states[b * 10 + 9];              // last column of (B,10)
}

__global__ void extract_col0_kernel(const float* __restrict__ W, float* __restrict__ dst,
                                    int n, int stride) {
    int j = blockIdx.x * blockDim.x + threadIdx.x;
    if (j < n) dst[j] = W[(size_t)j * stride];
}

// Pack weight -> WMMA B-fragment layout (bf16).  Logical B is K x N.
// transposed=1: B[k][n] = W[n*stride + k]; transposed=0: B[k][n] = W[k*stride + n]
// B-frag layout (32x16): lane&15 = n, k = kt*32 + (lane>>4)*8 + (e<8 ? e : 8+e)
__global__ void pack_b_kernel(const float* __restrict__ W, bf16_t* __restrict__ dst,
                              int NKT, int stride, int transposed, int total) {
    int id = blockIdx.x * blockDim.x + threadIdx.x;
    if (id >= total) return;
    int e    = id & 15;
    int lane = (id >> 4) & 31;
    int tile = id >> 9;             // nt*NKT + kt
    int kt   = tile % NKT;
    int nt   = tile / NKT;
    int k = kt * 32 + ((lane >> 4) << 3) + (e < 8 ? e : 8 + e);
    int n = nt * 16 + (lane & 15);
    float v = transposed ? W[(size_t)n * stride + k]
                         : W[(size_t)k * stride + n];
    dst[id] = (bf16_t)v;
}

// Pack f32 activation matrix (M x K, row-major, ld=ldA) -> packed A-fragments.
__global__ void pack_a_kernel(const float* __restrict__ A, bf16_t* __restrict__ dst,
                              int NKT, int ldA, int total) {
    int id = blockIdx.x * blockDim.x + threadIdx.x;
    if (id >= total) return;
    int e    = id & 15;
    int lane = (id >> 4) & 31;
    int tile = id >> 9;             // mt*NKT + kt
    int kt   = tile % NKT;
    int mt   = tile / NKT;
    int m  = mt * 16 + (lane & 15);
    int kh = (lane >> 4) << 3;
    int k  = kt * 32 + kh + (e < 8 ? e : 8 + e);
    dst[id] = (bf16_t)A[(size_t)m * ldA + k];
}

// ---------------------------------------------------------------- WMMA GEMM
// A operands are packed bf16 fragments; whole block's A K-range staged in LDS.
// Block: 128 threads = 4 waves. Wave computes 32(M) x 64(N); block = 32 x 256.
// Grid: (M/32, N/256).
// C = A0*B0 [+ A1*B1] + bias[col] + addend[row][col] + rowScale[row]*rowW[col]; opt gelu.
// Output: f32 row-major (Cf32) or packed-A bf16 (Cpack with packNKT = N/32).
__global__ void __launch_bounds__(128)
gemm_wmma_bf16(const bf16_t* __restrict__ A0p, int NKT0,
               const bf16_t* __restrict__ A1p, int NKT1,
               const bf16_t* __restrict__ B0p, const bf16_t* __restrict__ B1p,
               float* __restrict__ Cf32, int ldC,
               bf16_t* __restrict__ Cpack, int packNKT,
               const float* __restrict__ bias,
               const float* __restrict__ addend, int ldAdd,
               const float* __restrict__ rowScale, const float* __restrict__ rowW,
               int do_gelu) {
    __shared__ __align__(32) bf16_t As[32768];   // 64 KB: A0 at 0, A1 at 16384

    int lane = threadIdx.x & 31;
    int wave = threadIdx.x >> 5;
    int nt0  = blockIdx.y * 16 + wave * 4;

    // cooperative stage of this block's two M-tiles (contiguous in packed layout)
    {
        const uint4* src = (const uint4*)(A0p + (size_t)blockIdx.x * 2 * NKT0 * 512);
        uint4* dst = (uint4*)As;
        int nchunks = 128 * NKT0;                 // (2*NKT0*512)/8 16B-chunks
        for (int i = threadIdx.x; i < nchunks; i += 128) dst[i] = src[i];
        if (A1p != nullptr) {
            const uint4* src1 = (const uint4*)(A1p + (size_t)blockIdx.x * 2 * NKT1 * 512);
            uint4* dst1 = (uint4*)(As + 16384);
            int n1 = 128 * NKT1;
            for (int i = threadIdx.x; i < n1; i += 128) dst1[i] = src1[i];
        }
        __syncthreads();
    }

    v8f acc0[4] = {};
    v8f acc1[4] = {};

    {
        int nkt = NKT0;
        for (int kt = 0; kt < nkt; ++kt) {
            v16bf a0 = *(const v16bf*)(As + ((size_t)kt * 32 + lane) * 16);
            v16bf a1 = *(const v16bf*)(As + (((size_t)(nkt + kt)) * 32 + lane) * 16);
#pragma unroll
            for (int t = 0; t < 4; ++t) {
                const bf16_t* bp = B0p + ((((size_t)(nt0 + t)) * nkt + kt) * 32 + lane) * 16;
                v16bf b = *(const v16bf*)bp;
                acc0[t] = __builtin_amdgcn_wmma_f32_16x16x32_bf16(
                    false, a0, false, b, (short)0, acc0[t], false, false);
                acc1[t] = __builtin_amdgcn_wmma_f32_16x16x32_bf16(
                    false, a1, false, b, (short)0, acc1[t], false, false);
            }
        }
    }
    if (A1p != nullptr) {
        int nkt = NKT1;
        const bf16_t* As1 = As + 16384;
        for (int kt = 0; kt < nkt; ++kt) {
            v16bf a0 = *(const v16bf*)(As1 + ((size_t)kt * 32 + lane) * 16);
            v16bf a1 = *(const v16bf*)(As1 + (((size_t)(nkt + kt)) * 32 + lane) * 16);
#pragma unroll
            for (int t = 0; t < 4; ++t) {
                const bf16_t* bp = B1p + ((((size_t)(nt0 + t)) * nkt + kt) * 32 + lane) * 16;
                v16bf b = *(const v16bf*)bp;
                acc0[t] = __builtin_amdgcn_wmma_f32_16x16x32_bf16(
                    false, a0, false, b, (short)0, acc0[t], false, false);
                acc1[t] = __builtin_amdgcn_wmma_f32_16x16x32_bf16(
                    false, a1, false, b, (short)0, acc1[t], false, false);
            }
        }
    }

    // C/D layout: VGPR r, lanes 0-15 -> M=r, N=lane; lanes 16-31 -> M=8+r, N=lane-16
    int n  = lane & 15;
    int mr = (lane >> 4) << 3;
#pragma unroll
    for (int mt = 0; mt < 2; ++mt) {
        v8f* acc = (mt == 0) ? acc0 : acc1;
#pragma unroll
        for (int t = 0; t < 4; ++t) {
            int col = (nt0 + t) * 16 + n;
            float bv  = bias ? bias[col] : 0.0f;
            float rwv = rowW ? rowW[col] : 0.0f;
#pragma unroll
            for (int r = 0; r < 8; ++r) {
                int row = blockIdx.x * 32 + mt * 16 + mr + r;
                float v = acc[t][r] + bv;
                if (addend)   v += addend[(size_t)row * ldAdd + col];
                if (rowScale) v += rowScale[row] * rwv;
                if (do_gelu)  v = gelu_tanh_(v);
                if (Cpack) Cpack[packa_idx(row, col, packNKT)] = (bf16_t)v;
                else       Cf32[(size_t)row * ldC + col] = v;
            }
        }
    }
}

// ---------------------------------------------------------------- per-step kernels
// Reads gates, updates c (f32) and writes h directly into packed-A bf16 (NKT=16).
__global__ void lstm_update_kernel(const float* __restrict__ gates,
                                   bf16_t* __restrict__ hpack, float* __restrict__ c) {
    int i = blockIdx.x * blockDim.x + threadIdx.x;
    if (i >= BATCH * HID) return;
    int b = i >> 9;            // /512
    int j = i & 511;
    const float* g = gates + (size_t)b * GDIM;
    float ig = sigmoidf_(g[j]);
    float fg = sigmoidf_(g[512 + j]);
    float gg = tanhf(g[1024 + j]);
    float og = sigmoidf_(g[1536 + j]);
    float cn = fg * c[i] + ig * gg;
    c[i] = cn;
    float hv = og * tanhf(cn);
    hpack[packa_idx(b, j, HID / 32)] = (bf16_t)hv;
}

__global__ void predcorr_kernel(const float* __restrict__ s, const float* __restrict__ E,
                                const float* __restrict__ rate,
                                const float* __restrict__ cW1, const float* __restrict__ cb1,
                                const float* __restrict__ cW2, const float* __restrict__ cb2,
                                float* __restrict__ pred,
                                float* __restrict__ gens, float* __restrict__ bps, int t) {
    int b = blockIdx.x * blockDim.x + threadIdx.x;
    if (b >= BATCH) return;
    float sv = s[b], e = E[b];
    float corr = cb2[0];
#pragma unroll
    for (int k = 0; k < 32; ++k) {
        float hk = sv * cW1[k] + e * cW1[32 + k] + cb1[k];
        corr += tanhf(hk) * cW2[k];
    }
    float gen = rate[b] * (1.0f - sv);
    float p = sv + gen + corr;            // DT = 1
    float bp = 1.0f / (1.0f + __expf(-10.0f * (p - 0.5f)));
    pred[b] = p;
    gens[b * TLEN + t] = gen;
    bps[b * TLEN + t] = bp;
}

__global__ void ffn3_update_kernel(const float* __restrict__ f2,
                                   const float* __restrict__ fW3, const float* __restrict__ fb3,
                                   float* __restrict__ s, float* __restrict__ preds, int t) {
    int b = blockIdx.x * blockDim.x + threadIdx.x;
    if (b >= BATCH) return;
    const float* f = f2 + (size_t)b * FFH;
    float acc = fb3[0];
    for (int k = 0; k < FFH; ++k) acc += f[k] * fW3[k];
    float sv = s[b];
    float nxt = acc > sv ? acc : sv;
    nxt = fminf(fmaxf(nxt, 0.0f), 1.0f);
    preds[b * TLEN + t] = nxt;
    s[b] = nxt;
}

// ---------------------------------------------------------------- host
#define ALIGN_UP(x) (((x) + 255) & ~(size_t)255)

extern "C" void kernel_launch(void* const* d_in, const int* in_sizes, int n_in,
                              void* d_out, int out_size, void* d_ws, size_t ws_size,
                              hipStream_t stream) {
    const float* init_states = (const float*)d_in[0];
    const float* latent      = (const float*)d_in[1];
    const float* voltage     = (const float*)d_in[2];
    const float* thickness   = (const float*)d_in[3];
    const float* trap        = (const float*)d_in[4];
    const float* W_ih0       = (const float*)d_in[5];   // (2048, 257)
    const float* W_hh0       = (const float*)d_in[6];   // (2048, 512)
    const float* b0          = (const float*)d_in[7];
    const float* W_ih1       = (const float*)d_in[8];   // (2048, 512)
    const float* W_hh1       = (const float*)d_in[9];   // (2048, 512)
    const float* b1          = (const float*)d_in[10];
    const float* fW1         = (const float*)d_in[11];  // (513, 256)
    const float* fb1         = (const float*)d_in[12];
    const float* fW2         = (const float*)d_in[13];  // (256, 256)
    const float* fb2         = (const float*)d_in[14];
    const float* fW3         = (const float*)d_in[15];  // (256, 1)
    const float* fb3         = (const float*)d_in[16];
    const float* cW1         = (const float*)d_in[17];  // (2, 32)
    const float* cb1         = (const float*)d_in[18];
    const float* cW2         = (const float*)d_in[19];  // (32, 1)
    const float* cb2         = (const float*)d_in[20];

    float* out   = (float*)d_out;
    float* preds = out;                      // (B, 32)
    float* Eout  = out + BATCH * TLEN;       // (B,)
    float* gens  = Eout + BATCH;             // (B, 32)
    float* bps   = gens + BATCH * TLEN;      // (B, 32)

    // workspace carve-up
    char* ws = (char*)d_ws;
    size_t off = 0;
    auto take = [&](size_t bytes) { char* p = ws + off; off = ALIGN_UP(off + bytes); return p; };

    bf16_t* Whh0p = (bf16_t*)take((size_t)GDIM * 512 * 2);
    bf16_t* Wih1p = (bf16_t*)take((size_t)GDIM * 512 * 2);
    bf16_t* Whh1p = (bf16_t*)take((size_t)GDIM * 512 * 2);
    bf16_t* Wlatp = (bf16_t*)take((size_t)GDIM * 256 * 2);
    bf16_t* fW1p  = (bf16_t*)take((size_t)FFH * 512 * 2);
    bf16_t* fW2p  = (bf16_t*)take((size_t)FFH * 256 * 2);
    bf16_t* latp  = (bf16_t*)take((size_t)BATCH * LAT * 2);
    float*  ws0   = (float*)take((size_t)GDIM * 4);
    float*  latb  = (float*)take((size_t)BATCH * GDIM * 4);
    float*  gates = (float*)take((size_t)BATCH * GDIM * 4);
    // zero-init region (contiguous): h0p, h1p (bf16 packed), c0, c1 (f32)
    bf16_t* h0p   = (bf16_t*)take((size_t)BATCH * HID * 2);
    bf16_t* h1p   = (bf16_t*)take((size_t)BATCH * HID * 2);
    float*  c0    = (float*)take((size_t)BATCH * HID * 4);
    float*  c1    = (float*)take((size_t)BATCH * HID * 4);
    bf16_t* f1p   = (bf16_t*)take((size_t)BATCH * FFH * 2);
    float*  f2    = (float*)take((size_t)BATCH * FFH * 4);
    float*  sbuf  = (float*)take((size_t)BATCH * 4);
    float*  rate  = (float*)take((size_t)BATCH * 4);
    float*  Ews   = (float*)take((size_t)BATCH * 4);
    float*  predb = (float*)take((size_t)BATCH * 4);
    (void)ws_size; (void)in_sizes; (void)n_in; (void)out_size;

    // ---- setup (every call; deterministic) ----
    {
        // zero h0p,h1p,c0,c1 (contiguous, bf16 zero == 0x0000)
        int n = (int)(((size_t)BATCH * HID * 2 * 2 + (size_t)BATCH * HID * 4 * 2) / 4);
        zero_f32_kernel<<<(n + 255) / 256, 256, 0, stream>>>((float*)h0p, n);
    }
    init_kernel<<<(BATCH + 255) / 256, 256, 0, stream>>>(
        init_states, voltage, thickness, trap, Ews, Eout, rate, sbuf);
    extract_col0_kernel<<<(GDIM + 255) / 256, 256, 0, stream>>>(W_ih0, ws0, GDIM, 257);

    // pack weights -> WMMA B-fragment layout
    {
        int tot = (GDIM / 16) * (512 / 32) * 512;   // 1,048,576
        pack_b_kernel<<<(tot + 255) / 256, 256, 0, stream>>>(W_hh0, Whh0p, 16, 512, 1, tot);
        pack_b_kernel<<<(tot + 255) / 256, 256, 0, stream>>>(W_ih1, Wih1p, 16, 512, 1, tot);
        pack_b_kernel<<<(tot + 255) / 256, 256, 0, stream>>>(W_hh1, Whh1p, 16, 512, 1, tot);
    }
    {
        int tot = (GDIM / 16) * (256 / 32) * 512;   // W_ih0[:,1:], K=256
        pack_b_kernel<<<(tot + 255) / 256, 256, 0, stream>>>(W_ih0 + 1, Wlatp, 8, 257, 1, tot);
    }
    {
        int tot = (FFH / 16) * (512 / 32) * 512;    // fW1[1:,:], direct K x N
        pack_b_kernel<<<(tot + 255) / 256, 256, 0, stream>>>(fW1 + 256, fW1p, 16, 256, 0, tot);
    }
    {
        int tot = (FFH / 16) * (256 / 32) * 512;    // fW2
        pack_b_kernel<<<(tot + 255) / 256, 256, 0, stream>>>(fW2, fW2p, 8, 256, 0, tot);
    }
    // pack latent -> A fragments (once)
    {
        int tot = (BATCH / 16) * (LAT / 32) * 512;  // 262,144
        pack_a_kernel<<<(tot + 255) / 256, 256, 0, stream>>>(latent, latp, LAT / 32, LAT, tot);
    }

    // latbuf = latent @ W_ih0[:,1:]^T + b0   (step-invariant)
    {
        dim3 g(BATCH / 32, GDIM / 256);
        gemm_wmma_bf16<<<g, 128, 0, stream>>>(
            latp, LAT / 32, nullptr, 0, Wlatp, nullptr,
            latb, GDIM, nullptr, 0,
            b0, nullptr, 0, nullptr, nullptr, 0);
    }

    dim3 gGate(BATCH / 32, GDIM / 256);
    dim3 gFfn (BATCH / 32, FFH / 256);
    int bhBlocks = (BATCH * HID + 255) / 256;

    // ---- 32 recurrent steps ----
    for (int t = 0; t < TLEN; ++t) {
        // physics predictor + corrector MLP (uses s before update)
        predcorr_kernel<<<(BATCH + 255) / 256, 256, 0, stream>>>(
            sbuf, Ews, rate, cW1, cb1, cW2, cb2, predb, gens, bps, t);

        // LSTM 0: gates = h0 @ W_hh0^T + latbuf + s * W_ih0[:,0]
        gemm_wmma_bf16<<<gGate, 128, 0, stream>>>(
            h0p, HID / 32, nullptr, 0, Whh0p, nullptr,
            gates, GDIM, nullptr, 0,
            nullptr, latb, GDIM, sbuf, ws0, 0);
        lstm_update_kernel<<<bhBlocks, 256, 0, stream>>>(gates, h0p, c0);

        // LSTM 1: gates = h0 @ W_ih1^T + h1 @ W_hh1^T + b1
        gemm_wmma_bf16<<<gGate, 128, 0, stream>>>(
            h0p, HID / 32, h1p, HID / 32, Wih1p, Whh1p,
            gates, GDIM, nullptr, 0,
            b1, nullptr, 0, nullptr, nullptr, 0);
        lstm_update_kernel<<<bhBlocks, 256, 0, stream>>>(gates, h1p, c1);

        // FFN1: f1 = gelu(h1 @ fW1[1:,:] + pred * fW1[0,:] + fb1)  -> packed A
        gemm_wmma_bf16<<<gFfn, 128, 0, stream>>>(
            h1p, HID / 32, nullptr, 0, fW1p, nullptr,
            nullptr, 0, f1p, FFH / 32,
            fb1, nullptr, 0, predb, fW1 /* row 0 of fW1 */, 1);

        // FFN2: f2 = gelu(f1 @ fW2 + fb2)  -> f32
        gemm_wmma_bf16<<<gFfn, 128, 0, stream>>>(
            f1p, FFH / 32, nullptr, 0, fW2p, nullptr,
            f2, FFH, nullptr, 0,
            fb2, nullptr, 0, nullptr, nullptr, 1);

        // FFN3 + state update: nxt = clip(max(s, f2 @ fW3 + fb3), 0, 1)
        ffn3_update_kernel<<<(BATCH + 255) / 256, 256, 0, stream>>>(
            f2, fW3, fb3, sbuf, preds, t);
    }
}
